// DRML_net_33062658245170
// MI455X (gfx1250) — matrix operand order repaired and checked by reading, not compiled, not run on hardware
//
#include <hip/hip_runtime.h>
#include <cstdint>

// ---------------------------------------------------------------------------
// DRML net forward, MI455X (gfx1250), wave32.
//  * All GEMM-like work -> v_wmma_f32_16x16x32_f16 (f16 in, f32 accumulate).
//  * Conv input slabs + region patches DMA'd to LDS via the Tensor Data Mover
//    (tensor_load_to_lds, TENSORcnt) -- inline asm per CDNA5 ISA.
//  * Weights pre-packed in LDS as per-lane f16 A-fragments (ISA 7.12.2).
// Workspace requirement: (26,214,400 + 6,553,600) floats = 131,072,000 bytes.
// ---------------------------------------------------------------------------

typedef __attribute__((ext_vector_type(16))) _Float16 v16h;
typedef __attribute__((ext_vector_type(8)))  float    v8f;
typedef uint32_t u32x4 __attribute__((ext_vector_type(4)));
typedef uint32_t u32x8 __attribute__((ext_vector_type(8)));

union ABf16 { v16h h; _Float16 f[16]; uint32_t u[8]; };

__device__ __forceinline__ v8f wmma_f16(v16h a, v16h b, v8f c) {
  // 8 args: (neg_a, A, neg_b, B, c_mod, C, reuse_a, reuse_b)
  return __builtin_amdgcn_wmma_f32_16x16x32_f16(false, a, false, b, (short)0, c,
                                                false, false);
}

// A-matrix (16-bit, 16x32) per-lane K index for fragment element e (0..15):
//   half = (lane<16)?0:8 ;  K = e + half + (e<8 ? 0 : 8)
// B-matrix (16-bit, 32x16) per-lane K index: same as A, K = e + half + (e<8 ? 0 : 8)
// C/D (f32 16x16): reg r, lane l -> M = r + (l>=16?8:0), N = l&15

// ---------------------------------------------------------------------------
// Tensor Data Mover: async 3-D tile load Global -> LDS (f32 elements).
//   X: tile0 elements, contiguous        (tensor_dim0 = td0)
//   Y: tile1 rows, stride str0 elements  (tensor_dim1 = td1; OOB rows -> 0)
//   Z: tile2 planes, stride str1 elements(tensor_dim2 = td2)
// D# per CDNA5 ISA sect. 8; issued by one wave, tracked
// with TENSORcnt. workgroup_mask = 0 (not in a cluster).
// ---------------------------------------------------------------------------
__device__ __forceinline__ void tdm_load_tile_3d(
    const void* gptr, void* lptr,
    uint32_t td0, uint32_t td1, uint32_t td2,
    uint32_t tile0, uint32_t tile1, uint32_t tile2,
    uint32_t str0, uint32_t str1)
{
  uint64_t ga = (uint64_t)(uintptr_t)gptr;
  uint32_t la = (uint32_t)(uintptr_t)lptr; // generic LDS addr: low 32 = offset
  // group0: count=1 | lds_addr | global_addr[56:0] | type=2 ("image")
  u32x4 g0 = { 1u, la, (uint32_t)ga,
               ((uint32_t)(ga >> 32) & 0x01FFFFFFu) | (2u << 30) };
  // group1: mask=0, data_size=2(4B); dims/tiles/strides bit-packed
  u32x8 g1;
  g1[0] = 2u << 16;                               // data_size = 4 bytes
  g1[1] = (td0 & 0xFFFFu) << 16;                  // tensor_dim0[15:0]
  g1[2] = (td0 >> 16) | ((td1 & 0xFFFFu) << 16);  // td0[31:16] | td1[15:0]
  g1[3] = (td1 >> 16) | (tile0 << 16);            // td1[31:16] | tile_dim0
  g1[4] = tile1 | (tile2 << 16);                  // tile_dim1 | tile_dim2
  g1[5] = str0;                                   // tensor_dim0_stride[31:0]
  g1[6] = (str1 & 0xFFFFu) << 16;                 // str0[47:32]=0 | str1[15:0]
  g1[7] = str1 >> 16;                             // str1[47:16]
  u32x4 g2 = { td2, 0u, 0u, 0u };                 // tensor_dim2
  u32x4 g3 = { 0u, 0u, 0u, 0u };
  asm volatile("tensor_load_to_lds %0, %1, %2, %3"
               :: "s"(g0), "s"(g1), "s"(g2), "s"(g3)
               : "memory");
}

// ---------------------------------------------------------------------------
// Generic conv as WMMA: K dimension = (kernel position, padded channel).
// Input slab for the block's row group is TDM-loaded into LDS once.
// ---------------------------------------------------------------------------
template<int CIN, int CPAD, int COUT, int KH, int KW, int STRIDE, int RPB,
         bool RELU, int HIN, int WIN, int HOUT, int WOUT, int IN_BS, int OUT_BS>
__global__ __launch_bounds__(256) void conv_wmma(
    const float* __restrict__ in, const float* __restrict__ w,
    const float* __restrict__ bias, float* __restrict__ out)
{
  constexpr int MT     = COUT / 16;
  constexpr int XT     = (WOUT + 15) / 16;
  constexpr int POS    = KH * KW;
  constexpr int CHUNKS = (POS * CPAD + 31) / 32;
  constexpr int RB     = (HOUT + RPB - 1) / RPB;
  constexpr int SH     = (RPB - 1) * STRIDE + KH;   // slab rows

  __shared__ uint32_t wfrag[MT * CHUNKS * 256];   // [mt][chunk][lane][8] dwords
  __shared__ float    slab[CIN * SH * WIN];       // [c][slab_row][x]
  __shared__ float    sb[COUT];

  const int tid = threadIdx.x;
  const int b   = blockIdx.x / RB;
  const int rb  = blockIdx.x % RB;
  const int y0  = rb * RPB;

  // Kick off the async slab DMA (wave 0), overlap with weight packing.
  if (tid < 32) {
    tdm_load_tile_3d(in + (long)b * IN_BS + (long)(y0 * STRIDE) * WIN, slab,
                     (uint32_t)WIN, (uint32_t)(HIN - y0 * STRIDE), (uint32_t)CIN,
                     (uint32_t)WIN, (uint32_t)SH, (uint32_t)CIN,
                     (uint32_t)WIN, (uint32_t)(HIN * WIN));
  }

  // Pack weights into per-lane f16 A-fragments (zero-pad pos/channel overruns).
  for (int t = tid; t < MT * CHUNKS * 256; t += 256) {
    int i    = t & 7;
    int l    = (t >> 3) & 31;
    int rest = t >> 8;
    int chunk = rest % CHUNKS;
    int mt    = rest / CHUNKS;
    int half  = (l < 16) ? 0 : 8;
    int kl    = (i < 4) ? (half + 2 * i) : (16 + half + 2 * (i - 4));
    int o     = mt * 16 + (l & 15);
    _Float16 fv[2];
    #pragma unroll
    for (int j = 0; j < 2; ++j) {
      int kk  = chunk * 32 + kl + j;
      int pos = kk / CPAD;
      int c   = kk % CPAD;
      float v = 0.0f;
      if (pos < POS && c < CIN)
        v = w[((long)o * CIN + c) * POS + pos];
      fv[j] = (_Float16)v;
    }
    union { uint32_t u; _Float16 f[2]; } pk;
    pk.f[0] = fv[0]; pk.f[1] = fv[1];
    wfrag[t] = pk.u;
  }
  for (int t = tid; t < COUT; t += 256) sb[t] = bias[t];

  if (tid < 32) __builtin_amdgcn_s_wait_tensorcnt(0);  // slab DMA complete
  __syncthreads();

  const int wave = tid >> 5, lane = tid & 31;
  const int halfB = (lane < 16) ? 0 : 8;

  const int units = RPB * XT * MT;
  for (int u = wave; u < units; u += 8) {
    int ry  = u / (XT * MT);
    int rem = u % (XT * MT);
    int xt  = rem / MT;
    int mt  = rem % MT;
    int y   = y0 + ry;
    if (y >= HOUT) continue;
    int x0 = xt * 16;
    int n  = x0 + (lane & 15);
    int nx = (n < WOUT) ? n : (WOUT - 1);          // clamp addr, skip store

    v8f acc = {0.f,0.f,0.f,0.f,0.f,0.f,0.f,0.f};
    for (int chunk = 0; chunk < CHUNKS; ++chunk) {
      ABf16 a, bb;
      #pragma unroll
      for (int i = 0; i < 8; ++i)
        a.u[i] = wfrag[(mt * CHUNKS + chunk) * 256 + lane * 8 + i];
      #pragma unroll
      for (int e = 0; e < 16; ++e) {
        int kk  = chunk * 32 + halfB + e + ((e < 8) ? 0 : 8);
        int pos = kk / CPAD;
        int c   = kk % CPAD;
        if (pos >= POS) pos = POS - 1;              // weight is zero there
        if (c >= CIN)   c   = CIN - 1;              // weight is zero there
        int ky = pos / KW, kx = pos % KW;
        bb.f[e] = (_Float16)
            slab[c * (SH * WIN) + (ry * STRIDE + ky) * WIN + (nx * STRIDE + kx)];
      }
      acc = wmma_f16(a.h, bb.h, acc);
    }
    if (n < WOUT) {
      #pragma unroll
      for (int r = 0; r < 8; ++r) {
        int o = mt * 16 + r + ((lane >= 16) ? 8 : 0);
        float v = acc[r] + sb[o];
        if (RELU) v = fmaxf(v, 0.0f);
        out[(long)b * OUT_BS + (long)o * (HOUT * WOUT) + (long)y * WOUT + n] = v;
      }
    }
  }
}

// ---------------------------------------------------------------------------
// Fused region layer: per-(batch,patch) block.
// Patch is TDM-loaded to LDS; BN+ReLU into zero-padded LDS; 3x3 conv (32->32)
// via WMMA (K=288, 9 chunks); +bias +residual +ReLU overwriting the raw
// patch; 2x2 maxpool to h2.
// ---------------------------------------------------------------------------
__global__ __launch_bounds__(256) void region_kernel(
    const float* __restrict__ h1,
    const float* __restrict__ gamma, const float* __restrict__ beta,
    const float* __restrict__ mean,  const float* __restrict__ var,
    const float* __restrict__ cw,    const float* __restrict__ cb,
    float* __restrict__ h2)
{
  __shared__ float    xs[32 * 400];        // raw patch (residual), then output
  __shared__ float    xnp[32 * 22 * 22];   // padded BN/ReLU patch
  __shared__ uint32_t wfrag[2 * 9 * 256];  // [mt][chunk][lane][8]
  __shared__ float    sb[32];

  const int b  = blockIdx.x >> 6;
  const int pk = blockIdx.x & 63;
  const int pi = pk >> 3, pj = pk & 7;
  const int idx = pi * pj + pj;            // faithful to self.networks[i*j+j]
  const int tid = threadIdx.x;

  // Async patch DMA: [32 ch][20 rows][20 cols] from h1[b].
  if (tid < 32) {
    tdm_load_tile_3d(h1 + (long)b * 819200 + (pi * 20) * 160 + pj * 20, xs,
                     (uint32_t)(160 - pj * 20), (uint32_t)(160 - pi * 20), 32u,
                     20u, 20u, 32u,
                     160u, 25600u);
  }

  for (int t = tid; t < 32 * 484; t += 256) xnp[t] = 0.0f;

  const float* wsrc = cw + (long)idx * 9216;   // [32][32][3][3]
  for (int t = tid; t < 4608; t += 256) {
    int i = t & 7, l = (t >> 3) & 31, rest = t >> 8;
    int chunk = rest % 9, mt = rest / 9;
    int half = (l < 16) ? 0 : 8;
    int kl = (i < 4) ? (half + 2 * i) : (16 + half + 2 * (i - 4));
    int o  = mt * 16 + (l & 15);
    union { uint32_t u; _Float16 f[2]; } pk_;
    pk_.f[0] = (_Float16)wsrc[(o * 32 + kl)     * 9 + chunk];
    pk_.f[1] = (_Float16)wsrc[(o * 32 + kl + 1) * 9 + chunk];
    wfrag[t] = pk_.u;
  }
  if (tid < 32) sb[tid] = cb[idx * 32 + tid];

  if (tid < 32) __builtin_amdgcn_s_wait_tensorcnt(0);  // patch DMA complete
  __syncthreads();

  for (int t = tid; t < 12800; t += 256) {     // BN + ReLU -> padded patch
    int c = t / 400, p = t % 400, py = p / 20, px = p % 20;
    float g  = gamma[idx * 32 + c];
    float bt = beta [idx * 32 + c];
    float m  = mean [idx * 32 + c];
    float v  = var  [idx * 32 + c];
    float xn = (xs[t] - m) * (g * rsqrtf(v + 1e-5f)) + bt;
    xnp[c * 484 + (py + 1) * 22 + (px + 1)] = fmaxf(xn, 0.0f);
  }
  __syncthreads();

  const int wave = tid >> 5, lane = tid & 31;
  const int halfB = (lane < 16) ? 0 : 8;
  for (int u = wave; u < 50; u += 8) {       // 25 pixel tiles x 2 M tiles
    int pt = u >> 1, mt = u & 1;
    int p  = pt * 16 + (lane & 15);
    int py = p / 20, px = p % 20;
    v8f acc = {0.f,0.f,0.f,0.f,0.f,0.f,0.f,0.f};
    for (int chunk = 0; chunk < 9; ++chunk) {   // pos = chunk, K = channel
      ABf16 a, bb;
      #pragma unroll
      for (int i = 0; i < 8; ++i)
        a.u[i] = wfrag[(mt * 9 + chunk) * 256 + lane * 8 + i];
      int ky = chunk / 3, kx = chunk % 3;
      #pragma unroll
      for (int e = 0; e < 16; ++e) {
        int c = halfB + e + ((e < 8) ? 0 : 8);
        bb.f[e] = (_Float16)xnp[c * 484 + (py + ky) * 22 + (px + kx)];
      }
      acc = wmma_f16(a.h, bb.h, acc);
    }
    #pragma unroll
    for (int r = 0; r < 8; ++r) {            // +bias +residual, outer ReLU
      int o = mt * 16 + r + ((lane >= 16) ? 8 : 0);
      float v = acc[r] + sb[o] + xs[o * 400 + p];
      xs[o * 400 + p] = fmaxf(v, 0.0f);      // tiles partition (o,p): safe
    }
  }
  __syncthreads();

  for (int t = tid; t < 3200; t += 256) {    // 2x2 maxpool -> [B,32,80,80]
    int c = t / 100, q = t % 100, oy = q / 10, ox = q % 10;
    const float* r0 = &xs[c * 400 + (2 * oy) * 20 + 2 * ox];
    float m = fmaxf(fmaxf(r0[0], r0[1]), fmaxf(r0[20], r0[21]));
    h2[(long)b * 204800 + c * 6400 + (pi * 10 + oy) * 80 + (pj * 10 + ox)] = m;
  }
}

// ---------------------------------------------------------------------------
// FC as WMMA: M=batch(32 -> 2 tiles), N=features, K=inner.
// ---------------------------------------------------------------------------
template<bool RELU, bool TAIL>
__global__ __launch_bounds__(256) void fc_wmma(
    const float* __restrict__ act, int Kact, int K,
    const float* __restrict__ W, const float* __restrict__ bias,
    float* __restrict__ out, int Nout)
{
  int wave = (blockIdx.x * 256 + threadIdx.x) >> 5;
  int lane = threadIdx.x & 31;
  int ntiles = (Nout + 15) / 16;
  if (wave >= 2 * ntiles) return;
  int mt = wave & 1, nt = wave >> 1;
  int m0 = mt * 16, n0 = nt * 16;
  int halfA = (lane < 16) ? 0 : 8;
  int halfB = (lane < 16) ? 0 : 16;
  int bi = m0 + (lane & 15);
  int nr = n0 + (lane & 15); if (nr >= Nout) nr = Nout - 1;
  const float* arow = act + (long)bi * Kact;   // rows zero-padded past K
  const float* wrow = W   + (long)nr * K;
  int KC = (K + 31) / 32;

  v8f acc = {0.f,0.f,0.f,0.f,0.f,0.f,0.f,0.f};
  for (int chunk = 0; chunk < KC; ++chunk) {
    int k0 = chunk * 32;
    ABf16 a, bb;
    #pragma unroll
    for (int e = 0; e < 16; ++e) {
      int ka = k0 + halfA + e + ((e < 8) ? 0 : 8);
      a.f[e] = (_Float16)arow[ka];
      int kb = k0 + halfA + e + ((e < 8) ? 0 : 8);
      float v = TAIL ? ((kb < K) ? wrow[kb] : 0.0f) : wrow[kb];
      bb.f[e] = (_Float16)v;
    }
    acc = wmma_f16(a.h, bb.h, acc);
  }
  #pragma unroll
  for (int r = 0; r < 8; ++r) {
    int m  = m0 + r + ((lane >= 16) ? 8 : 0);
    int nn = n0 + (lane & 15);
    if (nn < Nout) {
      float v = acc[r] + bias[nn];
      if (RELU) v = fmaxf(v, 0.0f);
      out[(long)m * Nout + nn] = v;
    }
  }
}

// ---------------------------------------------------------------------------
// Small prep / epilogue kernels
// ---------------------------------------------------------------------------
__global__ void fold_bn_w(const float* __restrict__ w, const float* __restrict__ g,
                          const float* __restrict__ var, float* __restrict__ wout)
{ // conv2_w [16][32][64]: scale each input channel by gamma*rsqrt(var+eps)
  int t = blockIdx.x * 256 + threadIdx.x;
  if (t >= 32768) return;
  int i = (t >> 6) & 31;
  wout[t] = w[t] * (g[i] * rsqrtf(var[i] + 1e-5f));
}

__global__ void fold_bn_b(const float* __restrict__ w, const float* __restrict__ b0,
                          const float* __restrict__ g, const float* __restrict__ bt,
                          const float* __restrict__ mn, const float* __restrict__ var,
                          float* __restrict__ bout)
{
  int o = threadIdx.x;
  if (o >= 16) return;
  float acc = b0[o];
  for (int i = 0; i < 32; ++i) {
    float inv   = rsqrtf(var[i] + 1e-5f);
    float shift = bt[i] - mn[i] * g[i] * inv;
    float sw = 0.0f;
    for (int p = 0; p < 64; ++p) sw += w[(o * 32 + i) * 64 + p];
    acc += sw * shift;
  }
  bout[o] = acc;
}

__global__ void pad_zero(float* __restrict__ h6)
{ // zero the 16-float tail of each padded fc1 activation row (Kact=11680)
  int t = blockIdx.x * 256 + threadIdx.x;
  if (t >= 512) return;
  h6[(long)(t >> 4) * 11680 + 11664 + (t & 15)] = 0.0f;
}

__global__ void logsoftmax2(const float* __restrict__ f3, float* __restrict__ out)
{ // [32,24] -> log_softmax over axis 1 of [32,2,12]
  int t = blockIdx.x * 256 + threadIdx.x;
  if (t >= 384) return;
  int b = t / 12, a = t % 12;
  float h0 = f3[b * 24 + a], h1 = f3[b * 24 + 12 + a];
  float m = fmaxf(h0, h1);
  float l = m + logf(expf(h0 - m) + expf(h1 - m));
  out[b * 24 + a]      = h0 - l;
  out[b * 24 + 12 + a] = h1 - l;
}

// ---------------------------------------------------------------------------
extern "C" void kernel_launch(void* const* d_in, const int* in_sizes, int n_in,
                              void* d_out, int out_size, void* d_ws, size_t ws_size,
                              hipStream_t stream)
{
  (void)in_sizes; (void)n_in; (void)out_size; (void)ws_size;
  const float* x    = (const float*)d_in[0];
  const float* c1w  = (const float*)d_in[1];  const float* c1b = (const float*)d_in[2];
  const float* rlg  = (const float*)d_in[3];  const float* rlb = (const float*)d_in[4];
  const float* rlm  = (const float*)d_in[5];  const float* rlv = (const float*)d_in[6];
  const float* rlcw = (const float*)d_in[7];  const float* rlcb= (const float*)d_in[8];
  const float* bng  = (const float*)d_in[9];  const float* bnb = (const float*)d_in[10];
  const float* bnm  = (const float*)d_in[11]; const float* bnv = (const float*)d_in[12];
  const float* c2w  = (const float*)d_in[13]; const float* c2b = (const float*)d_in[14];
  const float* c3w  = (const float*)d_in[15]; const float* c3b = (const float*)d_in[16];
  const float* c4w  = (const float*)d_in[17]; const float* c4b = (const float*)d_in[18];
  const float* c5w  = (const float*)d_in[19]; const float* c5b = (const float*)d_in[20];
  const float* f1w  = (const float*)d_in[21]; const float* f1b = (const float*)d_in[22];
  const float* f2w  = (const float*)d_in[23]; const float* f2b = (const float*)d_in[24];
  const float* f3w  = (const float*)d_in[25]; const float* f3b = (const float*)d_in[26];

  float* ws  = (float*)d_ws;
  float* h1  = ws;              // [32,32,160,160]  (dead after region layer)
  float* h2  = ws + 26214400;   // [32,32,80,80]
  float* h3  = ws;              // [32,16,73,73]    (reuses h1 region)
  float* h4  = ws + 2728448;    // [32,16,66,66]
  float* h5  = ws + 4958720;    // [32,16,31,31]
  float* h6  = ws + 5450752;    // [32,11680] padded fc1 activations
  float* f1  = ws + 5824512;    // [32,4096]
  float* f2  = ws + 5955584;    // [32,2048]
  float* f3  = ws + 6021120;    // [32,24]
  float* w2f = ws + 6021888;    // BN-folded conv2 weights [16,32,8,8]
  float* b2f = ws + 6054656;    // BN-folded conv2 bias [16]
  float* out = (float*)d_out;

  // conv1: 3->32, 11x11, 170->160
  conv_wmma<3,4,32,11,11,1,1,false,170,170,160,160,86700,819200>
      <<<5120,256,0,stream>>>(x, c1w, c1b, h1);
  // region layer (+ReLU +maxpool fused) -> [32,32,80,80]
  region_kernel<<<2048,256,0,stream>>>(h1, rlg, rlb, rlm, rlv, rlcw, rlcb, h2);
  // fold post-pool BN into conv2 (exact: conv2 is valid-padded)
  fold_bn_w<<<128,256,0,stream>>>(c2w, bng, bnv, w2f);
  fold_bn_b<<<1,64,0,stream>>>(c2w, c2b, bng, bnb, bnm, bnv, b2f);
  // conv2: 32->16, 8x8, 80->73 (+ReLU)
  conv_wmma<32,32,16,8,8,1,2,true,80,80,73,73,204800,85264>
      <<<1184,256,0,stream>>>(h2, w2f, b2f, h3);
  // conv3: 16->16, 8x8, 73->66 (+ReLU)
  conv_wmma<16,16,16,8,8,1,2,true,73,73,66,66,85264,69696>
      <<<1056,256,0,stream>>>(h3, c3w, c3b, h4);
  // conv4: 16->16, 6x6, stride 2, 66->31 (+ReLU)
  conv_wmma<16,16,16,6,6,2,4,true,66,66,31,31,69696,15376>
      <<<256,256,0,stream>>>(h4, c4w, c4b, h5);
  // conv5: 16->16, 5x5, 31->27 (+ReLU), into K-padded fc1 activation rows
  pad_zero<<<2,256,0,stream>>>(h6);
  conv_wmma<16,16,16,5,5,1,4,true,31,31,27,27,15376,11680>
      <<<224,256,0,stream>>>(h5, c5w, c5b, h6);
  // FC stack
  fc_wmma<true,true ><<<64,256,0,stream>>>(h6, 11680, 11664, f1w, f1b, f1, 4096);
  fc_wmma<true,false><<<32,256,0,stream>>>(f1, 4096, 4096, f2w, f2b, f2, 2048);
  fc_wmma<false,false><<<1,256,0,stream>>>(f2, 2048, 2048, f3w, f3b, f3, 24);
  // log_softmax over the 2-way axis of [32,2,12]
  logsoftmax2<<<2,256,0,stream>>>(f3, out);
}
